// PNAModel_87205015978672
// MI455X (gfx1250) — matrix operand order repaired and checked by reading, not compiled, not run on hardware
//
#include <hip/hip_runtime.h>
#include <hip/hip_bf16.h>
#include <math.h>

// ---------------- problem constants ----------------
#define Hs    128
#define Ls    4
#define NGs   32
#define OUTs  12
#define NNODE 30000
#define NEDGE 480000
#define AVG_LOG 1.3821455f
#define EPS_BN  1e-5f
#define EPS_STD 1e-5f

typedef __bf16 bf16_t;
typedef __attribute__((ext_vector_type(16))) __bf16 v16bf;
typedef __attribute__((ext_vector_type(8)))  __bf16 v8bf;
typedef __attribute__((ext_vector_type(8)))  float  v8f;

// ---------------- CDNA5 helpers ----------------
__device__ inline v8f wmma_bf16(v16bf a, v16bf b, v8f c) {
  // D = A(16x32 bf16) x B(32x16 bf16) + C(16x16 f32)
  return __builtin_amdgcn_wmma_f32_16x16x32_bf16(
      false, a, false, b, (short)0, c, false, false);
}

// A fragment: lane m = lane&15, hi = lane>>4.
// hi=0 -> K = {kb..kb+7, kb+16..kb+23}; hi=1 -> K = {kb+8..15, kb+24..31}
__device__ inline v16bf load_afrag(const bf16_t* rowPtr, int kb, int hi) {
  v8bf a0 = *(const v8bf*)(rowPtr + kb + hi * 8);
  v8bf a1 = *(const v8bf*)(rowPtr + kb + 16 + hi * 8);
  v16bf r;
#pragma unroll
  for (int i = 0; i < 8; i++) { r[i] = a0[i]; r[8 + i] = a1[i]; }
  return r;
}

// B fragment from transposed weights WT[n][k]: lane n = lane&15 picks row,
// hi=0 -> K = {kb..kb+7, kb+16..kb+23}; hi=1 -> K = {kb+8..15, kb+24..31}
__device__ inline v16bf load_bfrag(const bf16_t* wtRow, int kb, int hi) {
  v8bf b0 = *(const v8bf*)(wtRow + kb + hi * 8);
  v8bf b1 = *(const v8bf*)(wtRow + kb + 16 + hi * 8);
  v16bf r;
#pragma unroll
  for (int i = 0; i < 8; i++) { r[i] = b0[i]; r[8 + i] = b1[i]; }
  return r;
}

// Software-pipelined 8-N-tile GEMM step over KTOT (multiple of 32).
// A row fixed per lane (aRow), B tiles at Bt[(t*16+nn)*ldB].
// Prefetches next fragment(s) before each WMMA so ds latency hides under WMMA.
#define PIPELINED_GEMM(aRow, Bt, ldB, KTOT, nn, hi, acc)                     \
  do {                                                                       \
    v16bf a_ = load_afrag((aRow), 0, (hi));                                  \
    v16bf b_ = load_bfrag(&(Bt)[(nn) * (ldB)], 0, (hi));                     \
    const int NIT_ = ((KTOT) / 32) * 8;                                      \
    _Pragma("unroll") for (int i_ = 0; i_ < NIT_; i_++) {                    \
      int t_ = i_ & 7;                                                       \
      v16bf ac_ = a_, bc_ = b_;                                              \
      if (i_ + 1 < NIT_) {                                                   \
        int kn_ = ((i_ + 1) >> 3) * 32, tn_ = (i_ + 1) & 7;                  \
        b_ = load_bfrag(&(Bt)[(tn_ * 16 + (nn)) * (ldB)], kn_, (hi));        \
        if (tn_ == 0) a_ = load_afrag((aRow), kn_, (hi));                    \
      }                                                                      \
      acc[t_] = wmma_bf16(ac_, bc_, acc[t_]);                                \
    }                                                                        \
  } while (0)

// native L2 float atomics (avoid CAS loops; CDNA5 has MIN_NUM/MAX_NUM f32)
__device__ inline void gatomic_add(float* p, float v) {
  asm volatile("global_atomic_add_f32 %0, %1, off" :: "v"(p), "v"(v) : "memory");
}
__device__ inline void gatomic_min(float* p, float v) {
  asm volatile("global_atomic_min_num_f32 %0, %1, off" :: "v"(p), "v"(v) : "memory");
}
__device__ inline void gatomic_max(float* p, float v) {
  asm volatile("global_atomic_max_num_f32 %0, %1, off" :: "v"(p), "v"(v) : "memory");
}

// CDNA5 async global->LDS copy (ASYNCcnt); 16 bytes per lane per op
__device__ inline void async_g2l_b128(unsigned lds_byte_off, const void* gaddr) {
  asm volatile("global_load_async_to_lds_b128 %0, %1, off"
               :: "v"(lds_byte_off), "v"(gaddr) : "memory");
}
__device__ inline void wait_asynccnt0() {
  asm volatile("s_wait_asynccnt 0x0" ::: "memory");
}

// ---------------- weight prep (tiny, f32 VALU) ----------------
// WT[n*ldWT+k] = (k<K) ? W[k*ldW+n] : 0   (transpose + bf16 + K zero-pad)
__global__ void k_wt_convert(const float* __restrict__ W, bf16_t* __restrict__ WT,
                             int K, int Kp, int Nn, int ldW, int ldWT) {
  int idx = blockIdx.x * blockDim.x + threadIdx.x;
  if (idx >= Nn * Kp) return;
  int n = idx / Kp, k = idx % Kp;
  float v = (k < K) ? W[k * ldW + n] : 0.0f;
  WT[n * ldWT + k] = (bf16_t)v;
}

// Wfold = eW @ W3 (128x128); WT[n][k]; bias[n] = eb@W3[:,n] + pb[n]
__global__ void k_fold_edge(const float* __restrict__ eW, const float* __restrict__ W3,
                            const float* __restrict__ eb, const float* __restrict__ pb,
                            bf16_t* __restrict__ WT, float* __restrict__ biasOut) {
  int idx = blockIdx.x * blockDim.x + threadIdx.x;
  if (idx >= Hs * Hs) return;
  int n = idx & (Hs - 1), k = idx >> 7;
  float s = 0.f;
  for (int j = 0; j < Hs; j++) s += eW[k * Hs + j] * W3[j * Hs + n];
  WT[n * Hs + k] = (bf16_t)s;
  if (k == 0) {
    float b = pb[n];
    for (int j = 0; j < Hs; j++) b += eb[j] * W3[j * Hs + n];
    biasOut[n] = b;
  }
}

// Wfold = post_W @ lin_W  (1664x128); WT[n][k]; bias[n] = post_b@linW[:,n]+lin_b[n]
__global__ void k_fold_post(const float* __restrict__ P, const float* __restrict__ Lw,
                            const float* __restrict__ pbias, const float* __restrict__ lbias,
                            bf16_t* __restrict__ WT, float* __restrict__ biasOut) {
  int idx = blockIdx.x * blockDim.x + threadIdx.x;
  if (idx >= 13 * Hs * Hs) return;
  int n = idx & (Hs - 1), k = idx >> 7;  // k < 1664
  float s = 0.f;
  for (int j = 0; j < Hs; j++) s += P[k * Hs + j] * Lw[j * Hs + n];
  WT[(size_t)n * (13 * Hs) + k] = (bf16_t)s;
  if (k == 0) {
    float b = lbias[n];
    for (int j = 0; j < Hs; j++) b += pbias[j] * Lw[j * Hs + n];
    biasOut[n] = b;
  }
}

// ---------------- small utility kernels ----------------
__global__ void k_fill(float* p, float v, int n) {
  int i = blockIdx.x * blockDim.x + threadIdx.x;
  if (i < n) p[i] = v;
}
__global__ void k_init_acc(float* s, float* q, float* mn, float* mx, int n) {
  int i = blockIdx.x * blockDim.x + threadIdx.x;
  if (i < n) { s[i] = 0.f; q[i] = 0.f; mn[i] = INFINITY; mx[i] = -INFINITY; }
}
__global__ void k_deg(const int* __restrict__ dst, float* __restrict__ deg, int E) {
  int i = blockIdx.x * blockDim.x + threadIdx.x;
  if (i < E) gatomic_add(deg + dst[i], 1.0f);
}

// ---------------- GEMM: f32 A (embeddings), C bf16 [M][128] ----------------
__global__ void k_gemm_emb(const float* __restrict__ A, int M, int K, int Kp, int ldA,
                           const bf16_t* __restrict__ WT /*[128][Kp]*/,
                           const float* __restrict__ bias,
                           bf16_t* __restrict__ C) {
  __shared__ __align__(32) bf16_t At[64 * 64];   // Kp <= 64 (8KB)
  __shared__ __align__(32) bf16_t Bt[128 * 64];  // (16KB)
  int tid = threadIdx.x;
  int rowBase = blockIdx.x * 64;
  for (int i = tid; i < 64 * Kp; i += 128) {
    int r = i / Kp, k = i % Kp;
    int gr = rowBase + r;
    float v = (gr < M && k < K) ? A[(size_t)gr * ldA + k] : 0.0f;
    At[r * Kp + k] = (bf16_t)v;
  }
  {  // stage weights: 128*Kp bf16 -> 16*Kp uint4 chunks
    const uint4* src = (const uint4*)WT;
    uint4* dst = (uint4*)Bt;
    for (int i = tid; i < 16 * Kp; i += 128) dst[i] = src[i];
  }
  __syncthreads();
  int wave = tid >> 5, lane = tid & 31, hi = lane >> 4, nn = lane & 15;
  int mloc = wave * 16 + (lane & 15);
  v8f acc[8] = {};
  if (Kp == 64) {
    PIPELINED_GEMM(&At[mloc * 64], Bt, 64, 64, nn, hi, acc);
  } else {
    PIPELINED_GEMM(&At[mloc * 32], Bt, 32, 32, nn, hi, acc);
  }
#pragma unroll
  for (int t = 0; t < 8; t++) {
    int col = t * 16 + nn;
#pragma unroll
    for (int r = 0; r < 8; r++) {
      int row = rowBase + wave * 16 + hi * 8 + r;
      if (row < M) C[(size_t)row * Hs + col] = (bf16_t)(acc[t][r] + bias[col]);
    }
  }
}

// ---------------- GEMM: ab = h @ [W1|W2]  (bf16 in/out, N=256) ----------------
__global__ void k_gemm_ab(const bf16_t* __restrict__ Ain, int M,
                          const bf16_t* __restrict__ WTab /*[256][128]*/,
                          bf16_t* __restrict__ ab /*[M][256]*/) {
  __shared__ __align__(32) bf16_t Bt[128 * 128];  // 32KB weight slice
  int rowBase = blockIdx.x * 64;
  const bf16_t* WT = WTab + (size_t)blockIdx.y * 128 * 128;
  int tid = threadIdx.x, wave = tid >> 5, lane = tid & 31, hi = lane >> 4, nn = lane & 15;
  {
    const uint4* src = (const uint4*)WT;
    uint4* dst = (uint4*)Bt;
    for (int i = tid; i < 2048; i += 128) dst[i] = src[i];
  }
  __syncthreads();
  int mrow = rowBase + wave * 16 + (lane & 15);
  if (mrow >= M) mrow = M - 1;  // safe clamp; stores guarded
  v8f acc[8] = {};
  PIPELINED_GEMM(Ain + (size_t)mrow * 128, Bt, 128, 128, nn, hi, acc);
#pragma unroll
  for (int t = 0; t < 8; t++) {
    int col = blockIdx.y * 128 + t * 16 + nn;
#pragma unroll
    for (int r = 0; r < 8; r++) {
      int row = rowBase + wave * 16 + hi * 8 + r;
      if (row < M) ab[(size_t)row * 256 + col] = (bf16_t)acc[t][r];
    }
  }
}

// ------- fused edge kernel: c = e_emb@Wfold_e; m = c + a[dst]+b[src]+bias;
// ------- atomic-aggregate sum/sumsq/min/max into per-dst node accumulators.
// ------- A tile streamed HBM->LDS with global_load_async_to_lds_b128;
// ------- B tile and gather buffer share one 32KB LDS region (B dead post-GEMM)
__global__ void k_edge_fused(const bf16_t* __restrict__ Eemb /*[E][128]*/,
                             const int* __restrict__ srcIdx, const int* __restrict__ dstIdx,
                             const bf16_t* __restrict__ WTe /*[128][128]*/,
                             const float* __restrict__ biasE,
                             const bf16_t* __restrict__ ab /*[N][256]*/,
                             float* __restrict__ aSum, float* __restrict__ aSq,
                             float* __restrict__ aMin, float* __restrict__ aMax) {
  __shared__ __align__(32) bf16_t At[64 * 128];            // 16KB e_emb tile
  __shared__ __align__(32) unsigned char uni[64 * 128 * 4]; // 32KB: Bt then gab
  __shared__ int sdst[64];
  bf16_t* Bt = (bf16_t*)uni;
  float* gab = (float*)uni;
  int tid = threadIdx.x;
  int ebase = blockIdx.x * 64;  // NEDGE % 64 == 0
  if (tid < 64) sdst[tid] = dstIdx[ebase + tid];
  // async-stage A tile: 16KB = 1024 x 16B chunks, 8 per thread (ASYNCcnt)
  {
    const bf16_t* gsrc = Eemb + (size_t)ebase * 128;
    for (int i = tid; i < 1024; i += 128) {
      unsigned lds_off = (unsigned)(uintptr_t)(&At[i * 8]);
      async_g2l_b128(lds_off, (const void*)(gsrc + i * 8));
    }
  }
  // stage B tile synchronously (L2-resident 32KB)
  {
    const uint4* src = (const uint4*)WTe;
    uint4* dst = (uint4*)Bt;
    for (int i = tid; i < 2048; i += 128) dst[i] = src[i];
  }
  wait_asynccnt0();
  __syncthreads();
  int wave = tid >> 5, lane = tid & 31, hi = lane >> 4, nn = lane & 15;
  int mloc = wave * 16 + (lane & 15);
  v8f acc[8] = {};
  PIPELINED_GEMM(&At[mloc * 128], Bt, 128, 128, nn, hi, acc);
  __syncthreads();  // B reads complete; reuse LDS region for gather buffer
  for (int i = tid; i < 64 * 128; i += 128) {
    int r = i >> 7, c = i & 127;
    int d = dstIdx[ebase + r], s = srcIdx[ebase + r];
    gab[i] = (float)ab[(size_t)d * 256 + c] + (float)ab[(size_t)s * 256 + 128 + c] + biasE[c];
  }
  __syncthreads();
#pragma unroll
  for (int t = 0; t < 8; t++) {
    int c = t * 16 + nn;
#pragma unroll
    for (int r = 0; r < 8; r++) {
      int lrow = wave * 16 + hi * 8 + r;
      float v = acc[t][r] + gab[lrow * 128 + c];
      size_t o = (size_t)sdst[lrow] * 128 + c;
      gatomic_add(aSum + o, v);
      gatomic_add(aSq + o, v * v);
      gatomic_min(aMin + o, v);
      gatomic_max(aMax + o, v);
    }
  }
}

// ------- aggregator finalize: mean/min/max/std + amp per node -------
__global__ void k_agg_finalize(const float* __restrict__ aSum, const float* __restrict__ aSq,
                               const float* __restrict__ aMin, const float* __restrict__ aMax,
                               const float* __restrict__ deg,
                               float* __restrict__ agg /*[N][512]*/, float* __restrict__ amp,
                               int Nn) {
  int idx = blockIdx.x * blockDim.x + threadIdx.x;
  if (idx >= Nn * 128) return;
  int n = idx >> 7, c = idx & 127;
  float dg = deg[n];
  float dc = fmaxf(dg, 1.0f);
  float mean = aSum[idx] / dc;
  float var = fmaxf(aSq[idx] / dc - mean * mean, 0.0f);
  float sd = sqrtf(var + EPS_STD);
  bool has = dg > 0.0f;
  float* o = agg + (size_t)n * 512;
  o[c] = mean;
  o[128 + c] = has ? aMin[idx] : 0.0f;
  o[256 + c] = has ? aMax[idx] : 0.0f;
  o[384 + c] = sd;
  if (c == 0) amp[n] = logf(dc + 1.0f) * (1.0f / AVG_LOG);
}

// ------- node post GEMM: [x | agg | agg*amp | agg/amp] (13H) @ Wfold_post
// ------- + bias; writes f32 pre-BN and block-reduced BN partial stats
__global__ void k_gemm_post(const bf16_t* __restrict__ h, const float* __restrict__ agg,
                            const float* __restrict__ amp, int M,
                            const bf16_t* __restrict__ WTp /*[128][1664]*/,
                            const float* __restrict__ biasP,
                            float* __restrict__ outF /*[M][128]*/,
                            float* __restrict__ bnS, float* __restrict__ bnQ) {
  __shared__ __align__(32) bf16_t At[64 * 128];   // 16KB
  __shared__ __align__(32) bf16_t Bt[128 * 128];  // 32KB weight K-chunk
  __shared__ float sS[128], sQ[128];
  int tid = threadIdx.x;
  if (tid < 128) { sS[tid] = 0.f; sQ[tid] = 0.f; }
  int rowBase = blockIdx.x * 64;
  int wave = tid >> 5, lane = tid & 31, hi = lane >> 4, nn = lane & 15;
  int mloc = wave * 16 + (lane & 15);
  v8f acc[8] = {};
  for (int kc = 0; kc < 13; kc++) {
    __syncthreads();
    int kbase = kc * 128;
    for (int i = tid; i < 64 * 128; i += 128) {  // build A chunk on the fly
      int r = i >> 7, c = i & 127;
      int gr = rowBase + r;
      float v = 0.0f;
      if (gr < M) {
        if (kc == 0) {
          v = (float)h[(size_t)gr * 128 + c];
        } else {
          int ai = (kc - 1) & 3, grp = (kc - 1) >> 2;
          float av = agg[(size_t)gr * 512 + ai * 128 + c];
          if (grp == 1) av *= amp[gr];
          else if (grp == 2) av /= amp[gr];
          v = av;
        }
      }
      At[i] = (bf16_t)v;
    }
    // stage this K-chunk of the folded weights: rows n=0..127, 256B each
    for (int i = tid; i < 2048; i += 128) {
      int n = i >> 4, q = i & 15;
      ((uint4*)Bt)[i] = *(const uint4*)(WTp + (size_t)n * 1664 + kbase + q * 8);
    }
    __syncthreads();
    PIPELINED_GEMM(&At[mloc * 128], Bt, 128, 128, nn, hi, acc);
  }
#pragma unroll
  for (int t = 0; t < 8; t++) {
    int c = t * 16 + nn;
    float bs = 0.f, bq = 0.f;
#pragma unroll
    for (int r = 0; r < 8; r++) {
      int row = rowBase + wave * 16 + hi * 8 + r;
      if (row < M) {
        float v = acc[t][r] + biasP[c];
        outF[(size_t)row * 128 + c] = v;
        bs += v; bq += v * v;
      }
    }
    atomicAdd(&sS[c], bs);   // ds_add_f32
    atomicAdd(&sQ[c], bq);
  }
  __syncthreads();
  if (tid < 128) { gatomic_add(bnS + tid, sS[tid]); gatomic_add(bnQ + tid, sQ[tid]); }
}

__global__ void k_bn_params(const float* bnS, const float* bnQ, float* mu, float* rstd, int Mn) {
  int c = threadIdx.x;
  if (c >= 128) return;
  float m = bnS[c] / (float)Mn;
  float v = bnQ[c] / (float)Mn - m * m;
  mu[c] = m;
  rstd[c] = rsqrtf(v + EPS_BN);
}

__global__ void k_bn_apply(float* __restrict__ hf, bf16_t* __restrict__ hb,
                           const float* __restrict__ mu, const float* __restrict__ rstd,
                           const float* __restrict__ gamma, const float* __restrict__ beta,
                           int Mn) {
  int idx = blockIdx.x * blockDim.x + threadIdx.x;
  if (idx >= Mn * 128) return;
  int c = idx & 127;
  float v = (hf[idx] - mu[c]) * rstd[c] * gamma[c] + beta[c];
  v = fmaxf(v, 0.0f);
  hf[idx] = v;
  hb[idx] = (bf16_t)v;
}

__global__ void k_pool(const float* __restrict__ hf, const int* __restrict__ batch,
                       float* __restrict__ g, int Mn) {
  int idx = blockIdx.x * blockDim.x + threadIdx.x;
  if (idx >= Mn * 128) return;
  int n = idx >> 7, c = idx & 127;
  gatomic_add(g + (size_t)batch[n] * 128 + c, hf[idx]);
}

__global__ void k_head(const float* __restrict__ g, const float* __restrict__ W1,
                       const float* __restrict__ b1, const float* __restrict__ W2,
                       const float* __restrict__ b2, float* __restrict__ out) {
  __shared__ float z[NGs * 64];
  int tid = threadIdx.x;
  for (int i = tid; i < NGs * 64; i += 128) {
    int r = i >> 6, c = i & 63;
    float s = b1[c];
    for (int k = 0; k < 128; k++) s += g[r * 128 + k] * W1[k * 64 + c];
    z[i] = fmaxf(s, 0.0f);
  }
  __syncthreads();
  for (int i = tid; i < NGs * OUTs; i += 128) {
    int r = i / OUTs, c = i % OUTs;
    float s = b2[c];
    for (int k = 0; k < 64; k++) s += z[r * 64 + k] * W2[k * OUTs + c];
    out[i] = s;
  }
}

// ---------------- orchestration ----------------
extern "C" void kernel_launch(void* const* d_in, const int* in_sizes, int n_in,
                              void* d_out, int out_size, void* d_ws, size_t ws_size,
                              hipStream_t stream) {
  // setup_inputs() dict order
  const float* x          = (const float*)d_in[0];
  const float* edge_attr  = (const float*)d_in[1];
  const int*   edge_index = (const int*)d_in[2];
  const int*   batch      = (const int*)d_in[3];
  const float* nemb_W = (const float*)d_in[4];
  const float* nemb_b = (const float*)d_in[5];
  const float* eemb_W = (const float*)d_in[6];
  const float* eemb_b = (const float*)d_in[7];
  const float* enc_W  = (const float*)d_in[8];
  const float* enc_b  = (const float*)d_in[9];
  const float* pre_W  = (const float*)d_in[10];
  const float* pre_b  = (const float*)d_in[11];
  const float* post_W = (const float*)d_in[12];
  const float* post_b = (const float*)d_in[13];
  const float* lin_W  = (const float*)d_in[14];
  const float* lin_b  = (const float*)d_in[15];
  const float* bn_g   = (const float*)d_in[16];
  const float* bn_b   = (const float*)d_in[17];
  const float* h1_W   = (const float*)d_in[18];
  const float* h1_b   = (const float*)d_in[19];
  const float* h2_W   = (const float*)d_in[20];
  const float* h2_b   = (const float*)d_in[21];
  float* out = (float*)d_out;

  const int* srcIdx = edge_index;          // edge_index[0]
  const int* dstIdx = edge_index + NEDGE;  // edge_index[1]

  // ---- workspace carve ----
  char* w = (char*)d_ws;
  auto alloc = [&](size_t bytes) -> void* {
    void* p = (void*)w;
    w += (bytes + 255) & ~(size_t)255;
    return p;
  };
  bf16_t* h_bf   = (bf16_t*)alloc((size_t)NNODE * Hs * 2);
  float*  h_f32  = (float*) alloc((size_t)NNODE * Hs * 4);
  bf16_t* ab     = (bf16_t*)alloc((size_t)NNODE * 256 * 2);
  bf16_t* e_emb  = (bf16_t*)alloc((size_t)NEDGE * Hs * 2);
  float*  aSum   = (float*) alloc((size_t)NNODE * Hs * 4);
  float*  aSq    = (float*) alloc((size_t)NNODE * Hs * 4);
  float*  aMin   = (float*) alloc((size_t)NNODE * Hs * 4);
  float*  aMax   = (float*) alloc((size_t)NNODE * Hs * 4);
  float*  agg    = (float*) alloc((size_t)NNODE * 512 * 4);
  float*  amp    = (float*) alloc((size_t)NNODE * 4);
  float*  deg    = (float*) alloc((size_t)NNODE * 4);
  float*  bnbuf  = (float*) alloc(512 * 4);   // S,Q,mu,rstd
  float*  gpool  = (float*) alloc((size_t)NGs * Hs * 4);
  bf16_t* WT_ne  = (bf16_t*)alloc((size_t)128 * 64 * 2);
  bf16_t* WT_ee  = (bf16_t*)alloc((size_t)128 * 32 * 2);
  bf16_t* WT_ab[Ls]; bf16_t* WT_e[Ls]; float* bias_e[Ls];
  bf16_t* WT_p[Ls]; float* bias_p[Ls];
  for (int l = 0; l < Ls; l++) {
    WT_ab[l]  = (bf16_t*)alloc((size_t)256 * 128 * 2);
    WT_e[l]   = (bf16_t*)alloc((size_t)128 * 128 * 2);
    bias_e[l] = (float*) alloc(128 * 4);
    WT_p[l]   = (bf16_t*)alloc((size_t)128 * 1664 * 2);
    bias_p[l] = (float*) alloc(128 * 4);
  }

  const int T = 256;
  auto g1 = [](int n, int t) { return (n + t - 1) / t; };

  // ---- weight prep (bf16, transposed, folded) ----
  k_wt_convert<<<g1(128 * 64, T), T, 0, stream>>>(nemb_W, WT_ne, 64, 64, 128, 128, 64);
  k_wt_convert<<<g1(128 * 32, T), T, 0, stream>>>(eemb_W, WT_ee, 16, 32, 128, 128, 32);
  for (int l = 0; l < Ls; l++) {
    const float* pW = pre_W + (size_t)l * 3 * Hs * Hs;
    k_wt_convert<<<g1(128 * 128, T), T, 0, stream>>>(pW, WT_ab[l], 128, 128, 128, 128, 128);
    k_wt_convert<<<g1(128 * 128, T), T, 0, stream>>>(pW + Hs * Hs, WT_ab[l] + 128 * 128,
                                                     128, 128, 128, 128, 128);
    k_fold_edge<<<g1(Hs * Hs, T), T, 0, stream>>>(enc_W + (size_t)l * Hs * Hs,
                                                  pW + 2 * Hs * Hs,
                                                  enc_b + l * Hs, pre_b + l * Hs,
                                                  WT_e[l], bias_e[l]);
    k_fold_post<<<g1(13 * Hs * Hs, T), T, 0, stream>>>(post_W + (size_t)l * 13 * Hs * Hs,
                                                       lin_W + (size_t)l * Hs * Hs,
                                                       post_b + l * Hs, lin_b + l * Hs,
                                                       WT_p[l], bias_p[l]);
  }

  // ---- embeddings + degree (degree is layer-invariant) ----
  k_gemm_emb<<<g1(NNODE, 64), 128, 0, stream>>>(x, NNODE, 64, 64, 64, WT_ne, nemb_b, h_bf);
  k_gemm_emb<<<g1(NEDGE, 64), 128, 0, stream>>>(edge_attr, NEDGE, 16, 32, 16, WT_ee, eemb_b, e_emb);
  k_fill<<<g1(NNODE, T), T, 0, stream>>>(deg, 0.0f, NNODE);
  k_deg<<<g1(NEDGE, T), T, 0, stream>>>(dstIdx, deg, NEDGE);

  // ---- PNA layers ----
  for (int l = 0; l < Ls; l++) {
    k_init_acc<<<g1(NNODE * Hs, T), T, 0, stream>>>(aSum, aSq, aMin, aMax, NNODE * Hs);
    k_gemm_ab<<<dim3(g1(NNODE, 64), 2), 128, 0, stream>>>(h_bf, NNODE, WT_ab[l], ab);
    k_edge_fused<<<NEDGE / 64, 128, 0, stream>>>(e_emb, srcIdx, dstIdx, WT_e[l], bias_e[l],
                                                 ab, aSum, aSq, aMin, aMax);
    k_agg_finalize<<<g1(NNODE * Hs, T), T, 0, stream>>>(aSum, aSq, aMin, aMax, deg,
                                                        agg, amp, NNODE);
    k_fill<<<1, T, 0, stream>>>(bnbuf, 0.0f, 256);
    k_gemm_post<<<g1(NNODE, 64), 128, 0, stream>>>(h_bf, agg, amp, NNODE, WT_p[l], bias_p[l],
                                                   h_f32, bnbuf, bnbuf + 128);
    k_bn_params<<<1, 128, 0, stream>>>(bnbuf, bnbuf + 128, bnbuf + 256, bnbuf + 384, NNODE);
    k_bn_apply<<<g1(NNODE * Hs, T), T, 0, stream>>>(h_f32, h_bf, bnbuf + 256, bnbuf + 384,
                                                    bn_g + l * Hs, bn_b + l * Hs, NNODE);
  }

  // ---- pool + head ----
  k_fill<<<1, T, 0, stream>>>(gpool, 0.0f, NGs * Hs);
  k_pool<<<g1(NNODE * Hs, T), T, 0, stream>>>(h_f32, batch, gpool, NNODE);
  k_head<<<1, 128, 0, stream>>>(gpool, h1_W, h1_b, h2_W, h2_b, out);
}
